// LinearHyperbolicPlusPlus_3685081940196
// MI455X (gfx1250) — matrix operand order repaired and checked by reading, not compiled, not run on hardware
//
#include <hip/hip_runtime.h>

typedef unsigned short u16;
typedef __attribute__((ext_vector_type(16))) __bf16 v16bf;
typedef __attribute__((ext_vector_type(8)))  float  v8f;
typedef __attribute__((ext_vector_type(4)))  int    v4i;

union Frag { v16bf v; uint4 q[2]; };

#define IN_DIM 1024
#define UNITS  1024
#define NROWS  16384
#define BM 128
#define BN 64
#define BK 64
#define LDK 72   /* padded LDS row stride (bf16 elems): 144B, keeps 16B alignment */

#ifndef __has_builtin
#define __has_builtin(x) 0
#endif
#if __has_builtin(__builtin_amdgcn_global_load_async_to_lds_b128) && \
    __has_builtin(__builtin_amdgcn_s_wait_asynccnt)
#define HAVE_ASYNC 1
#else
#define HAVE_ASYNC 0
#endif

static __device__ __forceinline__ u16 f2bf(float f) {
    unsigned int u = __float_as_uint(f);
    unsigned int r = (u + 0x7FFFu + ((u >> 16) & 1u)) >> 16;   // RNE
    return (u16)r;
}
static __device__ __forceinline__ float bf2f(u16 h) {
    return __uint_as_float(((unsigned int)h) << 16);
}

// 16-byte global -> LDS copy: async on CDNA5 (ASYNCcnt), sync fallback otherwise
#if HAVE_ASYNC
typedef __attribute__((address_space(1))) v4i g_v4i;   // global int4
typedef __attribute__((address_space(3))) v4i l_v4i;   // LDS int4
static __device__ __forceinline__ void cp16(const u16* g, u16* l) {
    __builtin_amdgcn_global_load_async_to_lds_b128(
        (g_v4i*)(uintptr_t)g,
        (l_v4i*)(unsigned)(uintptr_t)l, 0, 0);
}
#else
static __device__ __forceinline__ void cp16(const u16* g, u16* l) {
    *(uint4*)l = *(const uint4*)g;
}
#endif

// ---------------- Kernel A1: column norms of weigh_v + per-unit params ----------------
__global__ __launch_bounds__(256) void col_norms_kernel(
    const float* __restrict__ W, const float* __restrict__ wg,
    const float* __restrict__ bias, const float* __restrict__ cptr,
    float* __restrict__ vnorm, float* __restrict__ params)
{
    int j = blockIdx.x * blockDim.x + threadIdx.x;
    if (j >= UNITS) return;
    float s = 0.f;
    for (int i = 0; i < IN_DIM; ++i) {
        float v = W[(size_t)i * UNITS + j];
        s += v * v;
    }
    float n = sqrtf(s);
    n = fminf(fmaxf(n, 1e-6f), 1e6f);
    vnorm[j] = n;
    float rc = sqrtf(cptr[0]);
    float d  = 2.f * rc * bias[j];
    params[4 * j + 0] = coshf(d);
    params[4 * j + 1] = sinhf(d);
    params[4 * j + 2] = wg[j];
    params[4 * j + 3] = 0.f;
}

// ---------------- Kernel A2: zT[unit][i] = W[i][unit]/vnorm[unit], bf16 hi/lo split --
__global__ __launch_bounds__(256) void make_zT_kernel(
    const float* __restrict__ W, const float* __restrict__ vnorm,
    u16* __restrict__ Zhi, u16* __restrict__ Zlo)
{
    __shared__ float tile[32][33];
    int tx = threadIdx.x & 31;
    int ty = threadIdx.x >> 5;           // 0..7
    int i0 = blockIdx.y * 32;            // in_dim offset
    int j0 = blockIdx.x * 32;            // unit offset
    float nv = vnorm[j0 + tx];
#pragma unroll
    for (int k = 0; k < 4; ++k) {
        int ii = ty + k * 8;
        tile[ii][tx] = W[(size_t)(i0 + ii) * UNITS + j0 + tx] / nv;
    }
    __syncthreads();
#pragma unroll
    for (int k = 0; k < 4; ++k) {
        int u = ty + k * 8;
        float z = tile[tx][u];
        u16 hi = f2bf(z);
        u16 lo = f2bf(z - bf2f(hi));
        size_t o = (size_t)(j0 + u) * IN_DIM + i0 + tx;
        Zhi[o] = hi;
        Zlo[o] = lo;
    }
}

// ---------------- Kernel B: per-row expmap0+proj -> a = 2*rc*x (bf16 hi/lo) + cx2 ----
__global__ __launch_bounds__(256) void prep_x_kernel(
    const float* __restrict__ in, const float* __restrict__ cptr,
    u16* __restrict__ Ahi, u16* __restrict__ Alo, float* __restrict__ cx2)
{
    __shared__ float red[256];
    int m = blockIdx.x;
    int tid = threadIdx.x;
    const float* row = in + (size_t)m * IN_DIM;
    float s = 0.f;
    for (int j = tid; j < IN_DIM; j += 256) { float v = row[j]; s += v * v; }
    red[tid] = s;
    __syncthreads();
    for (int off = 128; off > 0; off >>= 1) {
        if (tid < off) red[tid] += red[tid + off];
        __syncthreads();
    }
    float sumsq = red[0];
    float rc = sqrtf(cptr[0]);
    float n  = sqrtf(fmaxf(sumsq, 1e-15f));
    float g0 = tanhf(rc * n) / (rc * n);                         // expmap0 factor
    float xn = sqrtf(fmaxf(g0 * g0 * sumsq, 1e-15f));            // safe_norm(x)
    float maxn = (1.f - 1e-5f) / rc;
    float g1 = (xn > maxn) ? (maxn / xn) : 1.f;                  // proj factor
    float fac = 2.f * rc * g0 * g1;
    if (tid == 0) {
        float rg = rc * g0 * g1;
        cx2[m] = rg * rg * sumsq;                                // sum((rc*x)^2)
    }
    for (int j = tid; j < IN_DIM; j += 256) {
        float a = fac * row[j];
        u16 hi = f2bf(a);
        u16 lo = f2bf(a - bf2f(hi));
        Ahi[(size_t)m * IN_DIM + j] = hi;
        Alo[(size_t)m * IN_DIM + j] = lo;
    }
}

// ---------------- Kernel C: bf16x3 WMMA GEMM, async-LDS double-buffered --------------
__global__ __launch_bounds__(256) void gemm_wmma_kernel(
    const u16* __restrict__ Ahi, const u16* __restrict__ Alo,
    const u16* __restrict__ Bhi, const u16* __restrict__ Blo,
    const float* __restrict__ cx2, const float* __restrict__ params,
    const float* __restrict__ cptr,
    float* __restrict__ Y, float* __restrict__ Sbuf, float* __restrict__ Pbuf)
{
    __shared__ __attribute__((aligned(16))) u16 sAhi[2][BM * LDK];
    __shared__ __attribute__((aligned(16))) u16 sAlo[2][BM * LDK];
    __shared__ __attribute__((aligned(16))) u16 sBhi[2][BN * LDK];
    __shared__ __attribute__((aligned(16))) u16 sBlo[2][BN * LDK];
    __shared__ float sS[BM];
    __shared__ float sP[BM];
    __shared__ float scx[BM];

    const int tid  = threadIdx.x;
    const int bm   = blockIdx.y * BM;
    const int bn   = blockIdx.x * BN;
    const int lane = tid & 31;
    const int wave = tid >> 5;
    const int wm   = (wave & 3) * 32;   // 4 waves along M
    const int wn   = (wave >> 2) * 32;  // 2 waves along N

    if (tid < BM) { sS[tid] = 0.f; sP[tid] = 0.f; scx[tid] = cx2[bm + tid]; }

    v8f acc[2][2];
#pragma unroll
    for (int a = 0; a < 2; ++a)
#pragma unroll
        for (int b = 0; b < 2; ++b) acc[a][b] = (v8f)(0.f);

    const int NK = IN_DIM / BK;   // 16 chunks

    // --- stage one K-chunk into LDS buffer `buf`: 12 async copies per wave ---
    auto stage = [&](int buf, int kc) {
#pragma unroll
        for (int it = 0; it < (BM * BK / 8) / 256; ++it) {   // 4 iters
            int idx = it * 256 + tid;
            int r  = idx >> 3;             // / (BK/8)
            int cq = (idx & 7) << 3;       // *8 bf16 = 16B
            size_t go = (size_t)(bm + r) * IN_DIM + kc + cq;
            int lo = r * LDK + cq;
            cp16(Ahi + go, &sAhi[buf][lo]);
            cp16(Alo + go, &sAlo[buf][lo]);
        }
#pragma unroll
        for (int it = 0; it < (BN * BK / 8) / 256; ++it) {   // 2 iters
            int idx = it * 256 + tid;
            int r  = idx >> 3;
            int cq = (idx & 7) << 3;
            size_t go = (size_t)(bn + r) * IN_DIM + kc + cq;
            int lo = r * LDK + cq;
            cp16(Bhi + go, &sBhi[buf][lo]);
            cp16(Blo + go, &sBlo[buf][lo]);
        }
    };

    stage(0, 0);                               // prologue: chunk 0 in flight

    for (int kt = 0; kt < NK; ++kt) {
        const int cur = kt & 1;
        if (kt + 1 < NK) {
            stage(cur ^ 1, (kt + 1) * BK);     // overlap next chunk with compute
#if HAVE_ASYNC
            __builtin_amdgcn_s_wait_asynccnt(12);   // chunk kt retired (in-order)
#endif
        } else {
#if HAVE_ASYNC
            __builtin_amdgcn_s_wait_asynccnt(0);
#endif
        }
        __syncthreads();                       // all waves' chunk-kt data visible

        const u16* pAhi = sAhi[cur];
        const u16* pAlo = sAlo[cur];
        const u16* pBhi = sBhi[cur];
        const u16* pBlo = sBlo[cur];

#pragma unroll
        for (int ks = 0; ks < BK; ks += 32) {
            // A frag (16-bit A 16x32): lanes 0-15: M=lane, K-base 0; lanes 16-31:
            // M=lane-16, K-base 8; elems 0..7 -> K=kb+0..7, 8..15 -> K=16+kb+0..7
            const int mr = (lane & 15);
            const int kb = (lane < 16) ? 0 : 8;
            Frag ah[2], al[2], bh[2], bl[2];
#pragma unroll
            for (int mf = 0; mf < 2; ++mf) {
                const u16* p = &pAhi[(wm + mf * 16 + mr) * LDK + ks + kb];
                ah[mf].q[0] = *(const uint4*)p;
                ah[mf].q[1] = *(const uint4*)(p + 16);
                const u16* q = &pAlo[(wm + mf * 16 + mr) * LDK + ks + kb];
                al[mf].q[0] = *(const uint4*)q;
                al[mf].q[1] = *(const uint4*)(q + 16);
            }
            // B frag (16-bit B 32x16): lane%16 = N; K-base 0 (lanes 0-15) / 8 (lanes 16-31);
            // elems 0..7 -> K=kb+0..7, 8..15 -> K=16+kb+0..7 (same as A)
            const int nc = (lane & 15);
            const int kboff = (lane < 16) ? 0 : 8;
#pragma unroll
            for (int nf = 0; nf < 2; ++nf) {
                const u16* p = &pBhi[(wn + nf * 16 + nc) * LDK + ks + kboff];
                bh[nf].q[0] = *(const uint4*)p;
                bh[nf].q[1] = *(const uint4*)(p + 16);
                const u16* q = &pBlo[(wn + nf * 16 + nc) * LDK + ks + kboff];
                bl[nf].q[0] = *(const uint4*)q;
                bl[nf].q[1] = *(const uint4*)(q + 16);
            }
#pragma unroll
            for (int mf = 0; mf < 2; ++mf)
#pragma unroll
                for (int nf = 0; nf < 2; ++nf) {
                    acc[mf][nf] = __builtin_amdgcn_wmma_f32_16x16x32_bf16(
                        false, ah[mf].v, false, bh[nf].v, (short)0, acc[mf][nf], false, false);
                    acc[mf][nf] = __builtin_amdgcn_wmma_f32_16x16x32_bf16(
                        false, ah[mf].v, false, bl[nf].v, (short)0, acc[mf][nf], false, false);
                    acc[mf][nf] = __builtin_amdgcn_wmma_f32_16x16x32_bf16(
                        false, al[mf].v, false, bh[nf].v, (short)0, acc[mf][nf], false, false);
                }
        }
        __syncthreads();   // protect buf `cur` before it is refilled at kt+2
    }

    // epilogue: mlr -> y, write y into Y(=d_out scratch), row sums S, P via LDS atomics
    const float rc = sqrtf(cptr[0]);
#pragma unroll
    for (int nf = 0; nf < 2; ++nf) {
        int n = bn + wn + nf * 16 + (lane & 15);
        float ch = params[4 * n + 0];
        float sh = params[4 * n + 1];
        float g  = params[4 * n + 2];
#pragma unroll
        for (int mf = 0; mf < 2; ++mf) {
            int mloc0 = wm + mf * 16;
#pragma unroll
            for (int r = 0; r < 8; ++r) {
                int mloc = mloc0 + ((lane < 16) ? r : r + 8);   // C/D: VGPR r -> M=r / r+8
                float dot = acc[mf][nf][r];
                float mlr = dot * ch - (1.f + scx[mloc]) * sh;
                float y = sinhf(2.f * g * asinhf(mlr)) / rc;
                Y[(size_t)(bm + mloc) * UNITS + n] = y;
                float y2 = y * y;
                atomicAdd(&sS[mloc], y2);
                if (y > 0.f) atomicAdd(&sP[mloc], y2);
            }
        }
    }
    __syncthreads();
    if (tid < BM) {
        atomicAdd(&Sbuf[bm + tid], sS[tid]);
        atomicAdd(&Pbuf[bm + tid], sP[tid]);
    }
}

// ---------------- Kernel D: per-row scalar chain + relu scaling, in place ------------
__global__ __launch_bounds__(256) void finalize_kernel(
    float* __restrict__ Y, const float* __restrict__ Sbuf,
    const float* __restrict__ Pbuf, const float* __restrict__ cptr)
{
    int m = blockIdx.x;
    int tid = threadIdx.x;
    float c  = cptr[0];
    float rc = sqrtf(c);
    float S  = Sbuf[m];
    float P  = Pbuf[m];
    float s  = 1.f / (1.f + sqrtf(1.f + c * S));            // x3 = s*y
    float n3 = sqrtf(fmaxf(s * s * S, 1e-15f));
    float a1 = fminf(rc * n3, 1.f - 1e-7f);
    float f1 = atanhf(a1) / (rc * n3);                      // logmap0
    float nu = sqrtf(fmaxf(f1 * f1 * s * s * P, 1e-15f));   // ||relu(u)||
    float f2 = tanhf(rc * nu) / (rc * nu);                  // expmap0
    float nv = f2 * nu;
    float maxn = (1.f - 1e-5f) / rc;
    float f3 = (nv > maxn) ? (maxn / nv) : 1.f;             // proj
    float nvp = fminf(nv, maxn);
    float nvc = sqrtf(fmaxf(nvp * nvp, 1e-15f));
    float a4 = fminf(rc * nvc, 1.f - 1e-7f);
    float f4 = atanhf(a4) / (rc * nvc);                     // final logmap0
    float F = f4 * f3 * f2 * f1 * s;
    for (int j = tid; j < UNITS; j += 256) {
        float y = Y[(size_t)m * UNITS + j];
        Y[(size_t)m * UNITS + j] = (y > 0.f) ? F * y : 0.f;
    }
}

// ---------------- host launch ---------------------------------------------------------
extern "C" void kernel_launch(void* const* d_in, const int* in_sizes, int n_in,
                              void* d_out, int out_size, void* d_ws, size_t ws_size,
                              hipStream_t stream) {
    (void)in_sizes; (void)n_in; (void)out_size; (void)ws_size;
    const float* inputs = (const float*)d_in[0];   // (16384,1024)
    const float* weighv = (const float*)d_in[1];   // (1024,1024)
    const float* wg     = (const float*)d_in[2];   // (1,1024)
    const float* bias   = (const float*)d_in[3];   // (1024,)
    const float* cptr   = (const float*)d_in[4];   // (1,)
    float* out = (float*)d_out;                    // (16384,1024) f32; also y scratch

    char* ws = (char*)d_ws;
    // workspace layout (needs ~68.2 MB)
    u16*   Ahi    = (u16*)  (ws + 0);                       // 32 MB
    u16*   Alo    = (u16*)  (ws + 33554432ull);             // 32 MB
    u16*   Zhi    = (u16*)  (ws + 67108864ull);             // 2 MB
    u16*   Zlo    = (u16*)  (ws + 69206016ull);             // 2 MB
    float* cx2    = (float*)(ws + 71303168ull);             // 64 KB
    float* vnorm  = (float*)(ws + 71368704ull);             // 4 KB
    float* params = (float*)(ws + 71372800ull);             // 16 KB
    float* Sbuf   = (float*)(ws + 71389184ull);             // 64 KB
    float* Pbuf   = (float*)(ws + 71454720ull);             // 64 KB

    col_norms_kernel<<<(UNITS + 255) / 256, 256, 0, stream>>>(
        weighv, wg, bias, cptr, vnorm, params);
    make_zT_kernel<<<dim3(UNITS / 32, IN_DIM / 32), 256, 0, stream>>>(
        weighv, vnorm, Zhi, Zlo);
    prep_x_kernel<<<NROWS, 256, 0, stream>>>(inputs, cptr, Ahi, Alo, cx2);

    (void)hipMemsetAsync(Sbuf, 0, NROWS * sizeof(float), stream);
    (void)hipMemsetAsync(Pbuf, 0, NROWS * sizeof(float), stream);

    gemm_wmma_kernel<<<dim3(UNITS / BN, NROWS / BM), 256, 0, stream>>>(
        Ahi, Alo, Zhi, Zlo, cx2, params, cptr, out, Sbuf, Pbuf);

    finalize_kernel<<<NROWS, 256, 0, stream>>>(out, Sbuf, Pbuf, cptr);
}
